// CNNToLSTMCustomInterleaving_53291954208995
// MI455X (gfx1250) — matrix operand order repaired and checked by reading, not compiled, not run on hardware
//
#include <hip/hip_runtime.h>
#include <hip/hip_bf16.h>
#include <math.h>

typedef __attribute__((ext_vector_type(16))) _Float16 v16h;
typedef __attribute__((ext_vector_type(8)))  float    v8f;

// ---------------- problem constants ----------------
constexpr int NB = 8;      // batch
constexpr int LL = 512;    // seq len
constexpr int EE = 128;    // embed dim
constexpr int T1 = 1024;   // nonzero scatter region (all indices < 1024)
constexpr int TOUT = 4096; // full scattered length
constexpr int DD = 256;    // interleaved feature dim (2E)
constexpr int HG = 512;    // 4*H gates

// ---------------- workspace layout (bytes) ----------------
constexpr size_t OFF_XM    = 0;                                   // (B,E,L) f32
constexpr size_t SZ_XM     = (size_t)NB*EE*LL*4;
constexpr size_t OFF_UP    = OFF_XM + SZ_XM;                      // (B,1024,256) f32
constexpr size_t SZ_STREAM = (size_t)NB*T1*DD*4;
constexpr size_t OFF_MID   = OFF_UP  + SZ_STREAM;
constexpr size_t OFF_LOW   = OFF_MID + SZ_STREAM;
constexpr size_t OFF_S     = OFF_LOW + SZ_STREAM;                 // 256x256 f32 (sum s s^T)
constexpr size_t OFF_MU    = OFF_S    + 256*256*4;                // 256 f32
constexpr size_t OFF_AEIG  = OFF_MU   + 1024;                     // 256x256 f32
constexpr size_t OFF_VEIG  = OFF_AEIG + 256*256*4;                // 256x256 f32
constexpr size_t OFF_COMPS = OFF_VEIG + 256*256*4;                // 256x128 f32
constexpr size_t OFF_MUC   = OFF_COMPS + 256*128*4;               // 128 f32 (mu@comps)
constexpr size_t OFF_BR    = OFF_MUC + 1024;
// per-branch block:
constexpr size_t BR_MMAT = 0;                         // 256x512 f16 (comps @ wih^T)
constexpr size_t BR_WHHT = BR_MMAT + 256*512*2;       // 128x512 f16 (whh^T)
constexpr size_t BR_RVEC = BR_WHHT + 128*512*2;       // 512 f32
constexpr size_t BR_HSUM = BR_RVEC + 512*4;           // 8x128 f32
constexpr size_t BR_XG   = BR_HSUM + 1024*4;          // (8,1024,512) f32
constexpr size_t BR_SIZE = BR_XG + (size_t)NB*T1*HG*4;
constexpr size_t OFF_ME  = OFF_BR + 3*BR_SIZE;        // 8x128 f32
constexpr size_t WS_NEED = OFF_ME + 1024*4;

// ---------------- WMMA fragment helpers (CDNA5 16x16x32 f16) ----------------
__device__ inline v8f wmma16(v16h a, v16h b, v8f c) {
  return __builtin_amdgcn_wmma_f32_16x16x32_f16(false, a, false, b, (short)0, c, false, false);
}
// A fragment (16x32 f16), source row-major f32, lda = ld
__device__ inline v16h load_a_f32(const float* p, int ld, int row0, int k0) {
  int l = threadIdx.x & 31; int m = l & 15; int hi = l >> 4;
  v16h a;
#pragma unroll
  for (int j = 0; j < 16; ++j) {
    int K = ((j >> 3) << 4) + (hi << 3) + (j & 7);
    a[j] = (_Float16)p[(size_t)(row0 + m) * ld + (k0 + K)];
  }
  return a;
}
// A fragment from transposed source: logical A[m][K] = p[(k0+K)*ld + row0+m]
__device__ inline v16h load_a_f32T(const float* p, int ld, int row0, int k0) {
  int l = threadIdx.x & 31; int m = l & 15; int hi = l >> 4;
  v16h a;
#pragma unroll
  for (int j = 0; j < 16; ++j) {
    int K = ((j >> 3) << 4) + (hi << 3) + (j & 7);
    a[j] = (_Float16)p[(size_t)(k0 + K) * ld + row0 + m];
  }
  return a;
}
// B fragment (32x16 f16), source row-major K x N
__device__ inline v16h load_b_f16(const _Float16* p, int ld, int k0, int col0) {
  int l = threadIdx.x & 31; int n = l & 15; int hi = l >> 4;
  v16h b;
#pragma unroll
  for (int j = 0; j < 16; ++j) {
    int K = ((j >> 3) << 4) + (hi << 3) + (j & 7);
    b[j] = p[(size_t)(k0 + K) * ld + col0 + n];
  }
  return b;
}
__device__ inline v16h load_b_f32(const float* p, int ld, int k0, int col0) {
  int l = threadIdx.x & 31; int n = l & 15; int hi = l >> 4;
  v16h b;
#pragma unroll
  for (int j = 0; j < 16; ++j) {
    int K = ((j >> 3) << 4) + (hi << 3) + (j & 7);
    b[j] = (_Float16)p[(size_t)(k0 + K) * ld + col0 + n];
  }
  return b;
}
__device__ inline float sigm(float x) { return 1.f / (1.f + __expf(-x)); }

// ---------------- kernels ----------------
__global__ void k_embed(const int* __restrict__ x, const float* __restrict__ ew,
                        float* __restrict__ xm) {
  int tid = blockIdx.x * blockDim.x + threadIdx.x; // B*E*L
  if (tid >= NB * EE * LL) return;
  int l = tid % LL; int e = (tid / LL) % EE; int b = tid / (LL * EE);
  int tok = x[b * LL + l];
  xm[((size_t)(b * EE) + e) * LL + l] = ew[(size_t)tok * EE + e];
}

__device__ float convdot(const float* xb, const float* w, const float* bias,
                         int e, int k, int start) {
  float s = 0.f;
  for (int ci = 0; ci < EE; ++ci) {
    const float* xr = xb + ci * LL;
    const float* wr = w + ((size_t)e * EE + ci) * k;
    for (int kk = 0; kk < k; ++kk) {
      int pos = start + kk;
      if (pos >= 0 && pos < LL) s += wr[kk] * xr[pos];
    }
  }
  return s + bias[e];
}

// Fused conv + inverse-scatter + interleave into the three streams (t < 1024)
__global__ void k_conv(const float* __restrict__ xm,
                       const float* w2, const float* b2, const float* w4, const float* b4,
                       const float* w3, const float* b3, const float* w6, const float* b6,
                       const float* w5, const float* b5,
                       float* __restrict__ up, float* __restrict__ mid, float* __restrict__ low) {
  int t = blockIdx.x % T1; int b = blockIdx.x / T1; int e = threadIdx.x;
  const float* xb = xm + (size_t)b * EE * LL;
  float v2 = 0.f, v4 = 0.f, v3 = 0.f, v6 = 0.f, v5 = 0.f;
  // stream t2: k=2 stride=1 pad=0; t in [1,1022] -> p=(t-1)/2
  if (t >= 1 && t <= 1022) v2 = convdot(xb, w2, b2, e, 2, (t - 1) >> 1);
  // stream t4: k=4 stride=2 pad=0; sets {4i+1,4i+3,4i+4,4i+6}, i in [0,254]
  { int r = t & 3, i;
    if (r == 1) i = (t - 1) >> 2; else if (r == 3) i = (t - 3) >> 2;
    else if (r == 0) i = (t >> 2) - 1; else i = (t - 6) / 4;
    if (t >= 1 && i >= 0 && i <= 254) v4 = convdot(xb, w4, b4, e, 4, 2 * i);
  }
  // stream t3: k=3 stride=3 pad=2; odd t -> c3 index i in [1,170], start 3i-2
  { int r = t % 6; int i = -1;
    if (r == 3) i = (t + 3) / 6; else if (r == 5) i = (t + 1) / 6; else if (r == 1) i = (t - 1) / 6;
    if (i >= 1 && i <= 170) v3 = convdot(xb, w3, b3, e, 3, 3 * i - 2);
  }
  // stream t6: k=6 stride=3 pad=2; all residues; i in [1,169]; base {1,2,4,6}->i=0
  { int r = t % 6; int i;
    if (r == 3) i = (t + 3) / 6; else if (r == 5) i = (t + 1) / 6; else if (r == 1) i = (t - 1) / 6;
    else if (r == 2) i = (t - 2) / 6; else if (r == 4) i = (t - 4) / 6; else i = t / 6 - 1;
    if (i >= 1 && i <= 169) v6 = convdot(xb, w6, b6, e, 6, 3 * i - 2);
    else if (t == 1 || t == 2 || t == 4 || t == 6) v6 = convdot(xb, w6, b6, e, 6, -2);
  }
  // stream t5: k=5 stride=3 pad=0; j in [0,168] -> c5 index j+1, start 3(j+1)
  { int r = t % 6; int j = -1;
    if (r == 1) j = (t - 1) / 6; else if (r == 3) j = (t - 3) / 6; else if (r == 5) j = (t - 5) / 6;
    else if (r == 0) j = t / 6 - 1; else if (r == 2) j = (t - 8) / 6;
    if (j >= 0 && j <= 168) v5 = convdot(xb, w5, b5, e, 5, 3 * (j + 1));
  }
  size_t ro = ((size_t)b * T1 + t) * DD;
  up[ro + 2 * e] = v2;  up[ro + 2 * e + 1] = v4;
  mid[ro + 2 * e] = v3; mid[ro + 2 * e + 1] = v6;
  low[ro + 2 * e] = 0.f; low[ro + 2 * e + 1] = v5;
}

__global__ void k_mu(const float* __restrict__ up, float* __restrict__ mu) {
  int d = threadIdx.x; // 256
  float s = 0.f;
  for (int r = 0; r < NB * T1; ++r) s += up[(size_t)r * DD + d];
  mu[d] = s / (float)(NB * TOUT);
}

// S = up^T up  (256x256, K=8192) via WMMA; one 16x16 tile per wave
__global__ void k_cov(const float* __restrict__ up, float* __restrict__ S) {
  int wg = blockIdx.x * 4 + (threadIdx.x >> 5); // 256 waves
  int ti = wg >> 4, tj = wg & 15;
  v8f acc = {};
  for (int k0 = 0; k0 < NB * T1; k0 += 32) {
    v16h a  = load_a_f32T(up, DD, ti * 16, k0);
    v16h bm = load_b_f32 (up, DD, k0, tj * 16);
    acc = wmma16(a, bm, acc);
  }
  int l = threadIdx.x & 31; int n = l & 15, hi = l >> 4;
#pragma unroll
  for (int r = 0; r < 8; ++r)
    S[(size_t)(ti * 16 + r + hi * 8) * DD + tj * 16 + n] = acc[r];
}

// Parallel two-sided Jacobi eigensolver, single workgroup, A/V in global scratch
__global__ __launch_bounds__(256, 1)
void k_eig(const float* __restrict__ S, const float* __restrict__ mu,
           float* __restrict__ A, float* __restrict__ V) {
  __shared__ float cS[128], sS[128];
  __shared__ int   pS[128], qS[128];
  int tid = threadIdx.x;
  for (int idx = tid; idx < 256 * 256; idx += 256) {
    int i = idx >> 8, j = idx & 255;
    A[idx] = (S[idx] - (float)(NB * TOUT) * mu[i] * mu[j]) * (1.f / (float)(NB * TOUT - 1));
    V[idx] = (i == j) ? 1.f : 0.f;
  }
  __threadfence(); __syncthreads();
  for (int sw = 0; sw < 8; ++sw) {
    for (int r = 0; r < 255; ++r) {
      if (tid < 128) {
        int p = (tid == 0) ? 0 : 1 + ((tid - 1 + r) % 255);
        int q = 1 + ((255 - tid - 1 + r) % 255);
        if (p > q) { int tmp = p; p = q; q = tmp; }
        float app = A[p * 256 + p], aqq = A[q * 256 + q], apq = A[p * 256 + q];
        float c, s;
        if (fabsf(apq) < 1e-20f) { c = 1.f; s = 0.f; }
        else {
          float th = (aqq - app) / (2.f * apq);
          float tt = copysignf(1.f / (fabsf(th) + sqrtf(th * th + 1.f)), th);
          c = 1.f / sqrtf(1.f + tt * tt); s = tt * c;
        }
        cS[tid] = c; sS[tid] = s; pS[tid] = p; qS[tid] = q;
      }
      __syncthreads();
      for (int wi = tid; wi < 128 * 256; wi += 256) {     // row rotations (J^T A)
        int k = wi >> 8, col = wi & 255;
        int p = pS[k], q = qS[k]; float c = cS[k], s = sS[k];
        float ap = A[p * 256 + col], aq = A[q * 256 + col];
        A[p * 256 + col] = c * ap - s * aq;
        A[q * 256 + col] = s * ap + c * aq;
      }
      __threadfence(); __syncthreads();
      for (int wi = tid; wi < 128 * 256; wi += 256) {     // col rotations (A J), V J
        int k = wi >> 8, row = wi & 255;
        int p = pS[k], q = qS[k]; float c = cS[k], s = sS[k];
        float ap = A[row * 256 + p], aq = A[row * 256 + q];
        A[row * 256 + p] = c * ap - s * aq;
        A[row * 256 + q] = s * ap + c * aq;
        float vp = V[row * 256 + p], vq = V[row * 256 + q];
        V[row * 256 + p] = c * vp - s * vq;
        V[row * 256 + q] = s * vp + c * vq;
      }
      __threadfence(); __syncthreads();
    }
  }
}

__global__ void k_comps(const float* __restrict__ A, const float* __restrict__ V,
                        float* __restrict__ comps) {
  __shared__ float lam[256];
  int d = threadIdx.x;
  lam[d] = A[(size_t)d * 256 + d];
  __syncthreads();
  float my = lam[d]; int rank = 0;
  for (int j = 0; j < 256; ++j) {
    float lj = lam[j];
    if (lj > my || (lj == my && j < d)) rank++;
  }
  if (rank < 128)
    for (int i = 0; i < 256; ++i) comps[(size_t)i * 128 + rank] = V[(size_t)i * 256 + d];
}

__global__ void k_muc(const float* __restrict__ mu, const float* __restrict__ comps,
                      float* __restrict__ muc) {
  int j = threadIdx.x; // 128
  float s = 0.f;
  for (int i = 0; i < 256; ++i) s += mu[i] * comps[(size_t)i * 128 + j];
  muc[j] = s;
}

// Mmat = comps(256x128) @ wih^T(128x512), stored f16 row-major 256x512
__global__ void k_prep(const float* __restrict__ comps, const float* __restrict__ wih,
                       _Float16* __restrict__ Mm) {
  int idx = blockIdx.x * blockDim.x + threadIdx.x; // 256*512
  int i = idx >> 9, n = idx & 511;
  float s = 0.f;
  for (int d = 0; d < 128; ++d) s += comps[(size_t)i * 128 + d] * wih[(size_t)n * 128 + d];
  Mm[(size_t)i * 512 + n] = (_Float16)s;
}

__global__ void k_rvec(const float* __restrict__ muc, const float* __restrict__ wih,
                       const float* __restrict__ bih, const float* __restrict__ bhh,
                       float* __restrict__ rvec) {
  int n = threadIdx.x; // 512
  float s = bih[n] + bhh[n];
  for (int d = 0; d < 128; ++d) s -= muc[d] * wih[(size_t)n * 128 + d];
  rvec[n] = s;
}

__global__ void k_whht(const float* __restrict__ whh, _Float16* __restrict__ whhT) {
  int idx = blockIdx.x * blockDim.x + threadIdx.x; // 128*512
  int k = idx >> 9, n = idx & 511;
  whhT[(size_t)k * 512 + n] = (_Float16)whh[(size_t)n * 128 + k];
}

// xg(8192x512) = stream(8192x256) @ Mmat(256x512) + rvec, one 16x16 tile per wave
__global__ void k_xg(const float* __restrict__ Sst, const _Float16* __restrict__ Mm,
                     const float* __restrict__ rvec, float* __restrict__ xg) {
  int wg = blockIdx.x * 4 + (threadIdx.x >> 5); // 16384 waves
  int ti = wg >> 5, tj = wg & 31;
  v8f acc = {};
  for (int k0 = 0; k0 < DD; k0 += 32) {
    v16h a  = load_a_f32(Sst, DD, ti * 16, k0);
    v16h bm = load_b_f16(Mm, HG, k0, tj * 16);
    acc = wmma16(a, bm, acc);
  }
  int l = threadIdx.x & 31; int n = l & 15, hi = l >> 4;
  int col = tj * 16 + n;
  float rv = rvec[col];
#pragma unroll
  for (int r = 0; r < 8; ++r)
    xg[(size_t)(ti * 16 + r + hi * 8) * HG + col] = acc[r] + rv;
}

// Persistent per-branch LSTM recurrence: B-fragments of whh^T resident in VGPRs
__global__ __launch_bounds__(512, 1) void k_lstm(char* __restrict__ ws) {
  char* base = ws + OFF_BR + (size_t)blockIdx.x * BR_SIZE;
  const _Float16* whhT = (const _Float16*)(base + BR_WHHT);
  const float* rvec = (const float*)(base + BR_RVEC);
  const float* xg   = (const float*)(base + BR_XG);
  float* hsum       = (float*)(base + BR_HSUM);

  __shared__ float hbuf[16 * 128];  // rows 8..15 stay zero (A padding)
  __shared__ float cbuf[8 * 128];
  __shared__ float gbuf[16 * 512];

  int tid = threadIdx.x; int lane = tid & 31; int wid = tid >> 5;
  int m = lane & 15, hi = lane >> 4;
  for (int i = tid; i < 16 * 128; i += 512) hbuf[i] = 0.f;
  for (int i = tid; i < 8 * 128; i += 512) cbuf[i] = 0.f;
  __syncthreads();

  v16h bfr[2][4]; // [ntile][ktile], held in registers for all 4096 steps
#pragma unroll
  for (int nt = 0; nt < 2; ++nt)
#pragma unroll
    for (int kt = 0; kt < 4; ++kt)
      bfr[nt][kt] = load_b_f16(whhT, HG, kt * 32, (wid * 2 + nt) * 16);

  float hs[2] = {0.f, 0.f};
  for (int t = 0; t < TOUT; ++t) {
    v16h af[4];
#pragma unroll
    for (int kt = 0; kt < 4; ++kt) {
#pragma unroll
      for (int j = 0; j < 16; ++j) {
        int K = ((j >> 3) << 4) + (hi << 3) + (j & 7);
        af[kt][j] = (_Float16)hbuf[m * 128 + kt * 32 + K];
      }
    }
#pragma unroll
    for (int nt = 0; nt < 2; ++nt) {
      v8f acc = {};
#pragma unroll
      for (int kt = 0; kt < 4; ++kt) acc = wmma16(af[kt], bfr[nt][kt], acc);
      int col = (wid * 2 + nt) * 16 + m;
#pragma unroll
      for (int r = 0; r < 8; ++r) gbuf[(r + hi * 8) * 512 + col] = acc[r];
    }
    __syncthreads();
#pragma unroll
    for (int q = 0; q < 2; ++q) {
      int flat = tid + 512 * q; int b = flat >> 7; int j = flat & 127;
      const float* xr = (t < T1) ? (xg + ((size_t)(b * T1 + t)) * HG) : rvec;
      float gi = gbuf[b * 512 + j]       + xr[j];
      float gf = gbuf[b * 512 + 128 + j] + xr[128 + j];
      float gg = gbuf[b * 512 + 256 + j] + xr[256 + j];
      float go = gbuf[b * 512 + 384 + j] + xr[384 + j];
      float cc = sigm(gf) * cbuf[flat] + sigm(gi) * tanhf(gg);
      float hh = sigm(go) * tanhf(cc);
      cbuf[flat] = cc;
      hbuf[b * 128 + j] = hh;
      hs[q] += hh;
    }
    __syncthreads();
  }
  hsum[tid]       = hs[0] * (1.f / (float)TOUT);
  hsum[tid + 512] = hs[1] * (1.f / (float)TOUT);
}

__global__ void k_me(const float* __restrict__ xm, float* __restrict__ me) {
  int idx = blockIdx.x * blockDim.x + threadIdx.x; // 1024
  if (idx >= NB * EE) return;
  const float* row = xm + (size_t)idx * LL;
  float s = 0.f;
  for (int l = 0; l < LL; ++l) s += row[l];
  me[idx] = s / (float)LL;
}

__global__ void k_head(const char* __restrict__ ws, const float* fw,
                       const float* fc1w, const float* fc1b,
                       const float* fc2w, const float* fc2b,
                       const float* fc3w, const float* fc3b,
                       float* __restrict__ out) {
  const float* u  = (const float*)(ws + OFF_BR + 0 * BR_SIZE + BR_HSUM);
  const float* mm = (const float*)(ws + OFF_BR + 1 * BR_SIZE + BR_HSUM);
  const float* lo = (const float*)(ws + OFF_BR + 2 * BR_SIZE + BR_HSUM);
  const float* me = (const float*)(ws + OFF_ME);
  __shared__ float fused[128]; __shared__ float h1[256]; __shared__ float h2[16];
  int tid = threadIdx.x;
  float f0 = fw[0], f1 = fw[1], f2 = fw[2], f3 = fw[3];
  for (int b = 0; b < NB; ++b) {
    if (tid < 128)
      fused[tid] = f0 * u[b * 128 + tid] + f1 * mm[b * 128 + tid]
                 + f2 * lo[b * 128 + tid] + f3 * me[b * 128 + tid];
    __syncthreads();
    { float a = fc1b[tid];
      for (int d = 0; d < 128; ++d) a += fc1w[(size_t)tid * 128 + d] * fused[d];
      h1[tid] = a * sigm(a); }
    __syncthreads();
    if (tid < 16) {
      float a = fc2b[tid];
      for (int d = 0; d < 256; ++d) a += fc2w[(size_t)tid * 256 + d] * h1[d];
      h2[tid] = fmaxf(a, 0.f);
    }
    __syncthreads();
    if (tid == 0) {
      float a = fc3b[0];
      for (int d = 0; d < 16; ++d) a += fc3w[d] * h2[d];
      out[b] = a;
    }
    __syncthreads();
  }
}

// ---------------- launcher ----------------
extern "C" void kernel_launch(void* const* d_in, const int* in_sizes, int n_in,
                              void* d_out, int out_size, void* d_ws, size_t ws_size,
                              hipStream_t stream) {
  (void)in_sizes; (void)n_in; (void)out_size;
  if (ws_size < WS_NEED) return;

  const int*   x  = (const int*)d_in[0];
  const float* ew = (const float*)d_in[1];
  const float* w2 = (const float*)d_in[2];  const float* b2 = (const float*)d_in[3];
  const float* w4 = (const float*)d_in[4];  const float* b4 = (const float*)d_in[5];
  const float* w3 = (const float*)d_in[6];  const float* b3 = (const float*)d_in[7];
  const float* w6 = (const float*)d_in[8];  const float* b6 = (const float*)d_in[9];
  const float* w5 = (const float*)d_in[10]; const float* b5 = (const float*)d_in[11];
  const float* wih[3] = {(const float*)d_in[12], (const float*)d_in[16], (const float*)d_in[20]};
  const float* whh[3] = {(const float*)d_in[13], (const float*)d_in[17], (const float*)d_in[21]};
  const float* bih[3] = {(const float*)d_in[14], (const float*)d_in[18], (const float*)d_in[22]};
  const float* bhh[3] = {(const float*)d_in[15], (const float*)d_in[19], (const float*)d_in[23]};
  const float* fusew = (const float*)d_in[24];
  const float* fc1w = (const float*)d_in[25]; const float* fc1b = (const float*)d_in[26];
  const float* fc2w = (const float*)d_in[27]; const float* fc2b = (const float*)d_in[28];
  const float* fc3w = (const float*)d_in[29]; const float* fc3b = (const float*)d_in[30];

  char* ws = (char*)d_ws;
  float* xm    = (float*)(ws + OFF_XM);
  float* up    = (float*)(ws + OFF_UP);
  float* mid   = (float*)(ws + OFF_MID);
  float* low   = (float*)(ws + OFF_LOW);
  float* Smat  = (float*)(ws + OFF_S);
  float* mu    = (float*)(ws + OFF_MU);
  float* Aeig  = (float*)(ws + OFF_AEIG);
  float* Veig  = (float*)(ws + OFF_VEIG);
  float* comps = (float*)(ws + OFF_COMPS);
  float* muc   = (float*)(ws + OFF_MUC);
  float* me    = (float*)(ws + OFF_ME);
  const float* streams[3] = {up, mid, low};

  k_embed<<<(NB * EE * LL + 255) / 256, 256, 0, stream>>>(x, ew, xm);
  k_conv<<<NB * T1, EE, 0, stream>>>(xm, w2, b2, w4, b4, w3, b3, w6, b6, w5, b5, up, mid, low);
  k_mu<<<1, 256, 0, stream>>>(up, mu);
  k_cov<<<64, 128, 0, stream>>>(up, Smat);
  k_eig<<<1, 256, 0, stream>>>(Smat, mu, Aeig, Veig);
  k_comps<<<1, 256, 0, stream>>>(Aeig, Veig, comps);
  k_muc<<<1, 128, 0, stream>>>(mu, comps, muc);

  for (int br = 0; br < 3; ++br) {
    char* bb = ws + OFF_BR + (size_t)br * BR_SIZE;
    _Float16* Mm   = (_Float16*)(bb + BR_MMAT);
    _Float16* whhT = (_Float16*)(bb + BR_WHHT);
    float* rvec    = (float*)(bb + BR_RVEC);
    float* xg      = (float*)(bb + BR_XG);
    k_prep<<<512, 256, 0, stream>>>(comps, wih[br], Mm);
    k_rvec<<<1, 512, 0, stream>>>(muc, wih[br], bih[br], bhh[br], rvec);
    k_whht<<<256, 256, 0, stream>>>(whh[br], whhT);
    k_xg<<<4096, 128, 0, stream>>>(streams[br], Mm, rvec, xg);
  }

  k_lstm<<<3, 512, 0, stream>>>(ws);
  k_me<<<4, 256, 0, stream>>>(xm, me);
  k_head<<<1, 256, 0, stream>>>(ws, fusew, fc1w, fc1b, fc2w, fc2b, fc3w, fc3b, (float*)d_out);
}
